// MLA_70394513981757
// MI455X (gfx1250) — matrix operand order repaired and checked by reading, not compiled, not run on hardware
//
#include <hip/hip_runtime.h>
#include <hip/hip_bf16.h>

#define B_ 2
#define S_ 2048
#define D_ 2048
#define H_ 16
#define NOPE_ 128
#define ROPE_ 64
#define V_ 128
#define C_ 512
#define QK_ 192
#define KEXT_ 576   // C_ + ROPE_
#define LDSTR 40    // padded LDS row stride (32 + 8) to avoid bank conflicts

typedef __attribute__((ext_vector_type(16))) __bf16    bf16x16;
typedef __attribute__((ext_vector_type(8)))  float     f32x8;
typedef __attribute__((ext_vector_type(8)))  unsigned  u32x8;
typedef __attribute__((ext_vector_type(4)))  int       v4i;
typedef __attribute__((address_space(1)))    v4i       v4i_g;   // global int4
typedef __attribute__((address_space(3)))    v4i       v4i_l;   // LDS int4

#ifndef __has_builtin
#define __has_builtin(x) 0
#endif
#if __has_builtin(__builtin_amdgcn_global_load_async_to_lds_b128)
#define ASYNC_LDS 1
#else
#define ASYNC_LDS 0
#endif

// ---------------- helpers ----------------

__device__ __forceinline__ unsigned short f2bf(float f) {
    union { float f; unsigned u; } v; v.f = f;
    unsigned r = v.u + 0x7FFFu + ((v.u >> 16) & 1u);   // RNE
    return (unsigned short)(r >> 16);
}
__device__ __forceinline__ float bf2f(unsigned short h) {
    union { unsigned u; float f; } v; v.u = ((unsigned)h) << 16;
    return v.f;
}

__device__ __forceinline__ f32x8 wmma_bf16(u32x8 a, u32x8 b, f32x8 c) {
    bf16x16 av = __builtin_bit_cast(bf16x16, a);
    bf16x16 bv = __builtin_bit_cast(bf16x16, b);
    return __builtin_amdgcn_wmma_f32_16x16x32_bf16(
        false, av, false, bv, (short)0, c, false, false);
}

// 16-byte global -> LDS copy (async on CDNA5, sync fallback otherwise)
__device__ __forceinline__ void cp16(unsigned short* dst, const unsigned short* src) {
#if ASYNC_LDS
    __builtin_amdgcn_global_load_async_to_lds_b128(
        (v4i_g*)src, (v4i_l*)dst, 0, 0);
#else
    *(uint4*)dst = *(const uint4*)src;
#endif
}
__device__ __forceinline__ void wait_async() {
#if ASYNC_LDS
#if __has_builtin(__builtin_amdgcn_s_wait_asynccnt)
    __builtin_amdgcn_s_wait_asynccnt(0);
#else
    asm volatile("s_wait_asynccnt 0x0" ::: "memory");
#endif
#endif
}

// A fragment: 16x32 bf16; lane l: m=l%16, kbase=(l/16)*8; v0..3=kb+0..7, v4..7=kb+16..23
__device__ __forceinline__ u32x8 load_frag_a(const unsigned short* a, int lda) {
    int lane = threadIdx.x & 31;
    int m  = lane & 15;
    int kb = (lane >> 4) << 3;
    const unsigned short* p = a + (long)m * lda + kb;
    u32x8 u;
#pragma unroll
    for (int j = 0; j < 4; ++j) u[j]     = *(const unsigned int*)(p + 2 * j);
#pragma unroll
    for (int j = 0; j < 4; ++j) u[4 + j] = *(const unsigned int*)(p + 16 + 2 * j);
    return u;
}

// B fragment: B = W^T where W is (N,K) row-major; lane l: n=l%16, kbase=(l/16)*8; v0..3=kb+0..7, v4..7=kb+16..23
__device__ __forceinline__ u32x8 load_frag_b(const unsigned short* w, int ldw) {
    int lane = threadIdx.x & 31;
    int n  = lane & 15;
    int kb = (lane >> 4) << 3;
    const unsigned short* p = w + (long)n * ldw + kb;
    u32x8 u;
#pragma unroll
    for (int j = 0; j < 4; ++j) u[j] = *(const unsigned int*)(p + 2 * j);
#pragma unroll
    for (int j = 0; j < 4; ++j) u[4 + j] = *(const unsigned int*)(p + 16 + 2 * j);
    return u;
}

// ---------------- pointwise kernels ----------------

__global__ void f2bf_kernel(const float* __restrict__ src, unsigned short* __restrict__ dst, int n) {
    for (int i = blockIdx.x * blockDim.x + threadIdx.x; i < n; i += gridDim.x * blockDim.x)
        dst[i] = f2bf(src[i]);
}

__global__ void split_wkvb_kernel(const float* __restrict__ wkvb,
                                  unsigned short* __restrict__ wnt,
                                  unsigned short* __restrict__ wvb) {
    int idx = blockIdx.x * blockDim.x + threadIdx.x;
    if (idx >= H_ * 256 * 512) return;
    int c = idx & 511;
    int r = (idx >> 9) & 255;
    int h = idx >> 17;
    unsigned short v = f2bf(wkvb[idx]);
    if (r < NOPE_) wnt[((long)(h * C_ + c)) * NOPE_ + r] = v;
    else           wvb[((long)(h * V_ + (r - NOPE_))) * C_ + c] = v;
}

__global__ void rope_q_kernel(const unsigned short* __restrict__ q,
                              const float* __restrict__ fcos, const float* __restrict__ fsin,
                              unsigned short* __restrict__ qext) {
    int idx = blockIdx.x * blockDim.x + threadIdx.x;
    if (idx >= B_ * S_ * H_ * (ROPE_ / 2)) return;
    int i = idx & 31;
    int h = (idx >> 5) & (H_ - 1);
    int s = (idx >> 9) & (S_ - 1);
    int b = idx >> 20;
    long qb = ((long)(b * S_ + s) * H_ + h) * QK_ + NOPE_ + 2 * i;
    float tr = bf2f(q[qb]), ti = bf2f(q[qb + 1]);
    float cv = fcos[s * 32 + i], sv = fsin[s * 32 + i];
    long ob = ((long)(b * H_ + h) * S_ + s) * KEXT_ + C_ + 2 * i;
    qext[ob]     = f2bf(tr * cv - ti * sv);
    qext[ob + 1] = f2bf(tr * sv + ti * cv);
}

__global__ __launch_bounds__(128) void kv_post_kernel(const float* __restrict__ kvf,
                                                      const float* __restrict__ fcos,
                                                      const float* __restrict__ fsin,
                                                      unsigned short* __restrict__ kb,
                                                      unsigned short* __restrict__ ct) {
    int rid = blockIdx.x;            // b*S + s
    int b = rid / S_, s = rid % S_;
    const float* src = kvf + (long)rid * KEXT_;
    __shared__ float red[128];
    float v[4]; float ss = 0.f;
#pragma unroll
    for (int i = 0; i < 4; ++i) { v[i] = src[threadIdx.x * 4 + i]; ss += v[i] * v[i]; }
    red[threadIdx.x] = ss; __syncthreads();
    for (int off = 64; off > 0; off >>= 1) {
        if ((int)threadIdx.x < off) red[threadIdx.x] += red[threadIdx.x + off];
        __syncthreads();
    }
    float inv = rsqrtf(red[0] / (float)C_ + 1e-6f);
#pragma unroll
    for (int i = 0; i < 4; ++i) {
        int c = threadIdx.x * 4 + i;
        unsigned short bv = f2bf(v[i] * inv);
        kb[(long)rid * KEXT_ + c] = bv;
        ct[((long)b * C_ + c) * S_ + s] = bv;
    }
    if (threadIdx.x < 32) {
        int i = threadIdx.x;
        float tr = src[C_ + 2 * i], ti = src[C_ + 2 * i + 1];
        float cv = fcos[s * 32 + i], sv = fsin[s * 32 + i];
        kb[(long)rid * KEXT_ + C_ + 2 * i]     = f2bf(tr * cv - ti * sv);
        kb[(long)rid * KEXT_ + C_ + 2 * i + 1] = f2bf(tr * sv + ti * cv);
    }
}

// ---------------- direct WMMA GEMM (small-N cases): Out = A * W^T ----------------
template <bool BF16OUT>
__global__ __launch_bounds__(128) void gemm_awt_kernel(
    const unsigned short* __restrict__ A, int lda,
    const unsigned short* __restrict__ W, int ldw,
    void* __restrict__ Out, int ldd, int K)
{
    int wave = threadIdx.x >> 5;
    int m0 = blockIdx.y * 64 + (wave >> 1) * 32;
    int n0 = blockIdx.x * 64 + (wave & 1) * 32;
    f32x8 acc00 = {}, acc01 = {}, acc10 = {}, acc11 = {};
    for (int k = 0; k < K; k += 32) {
        u32x8 a0 = load_frag_a(A + (long)m0 * lda + k, lda);
        u32x8 a1 = load_frag_a(A + (long)(m0 + 16) * lda + k, lda);
        u32x8 b0 = load_frag_b(W + (long)n0 * ldw + k, ldw);
        u32x8 b1 = load_frag_b(W + (long)(n0 + 16) * ldw + k, ldw);
        acc00 = wmma_bf16(a0, b0, acc00);
        acc01 = wmma_bf16(a0, b1, acc01);
        acc10 = wmma_bf16(a1, b0, acc10);
        acc11 = wmma_bf16(a1, b1, acc11);
    }
    int lane = threadIdx.x & 31;
    int n = lane & 15, hi = lane >> 4;
#pragma unroll
    for (int r = 0; r < 8; ++r) {
        int m = r + 8 * hi;
        if constexpr (BF16OUT) {
            unsigned short* o = (unsigned short*)Out;
            o[(long)(m0 + m) * ldd + n0 + n]           = f2bf(acc00[r]);
            o[(long)(m0 + m) * ldd + n0 + 16 + n]      = f2bf(acc01[r]);
            o[(long)(m0 + m + 16) * ldd + n0 + n]      = f2bf(acc10[r]);
            o[(long)(m0 + m + 16) * ldd + n0 + 16 + n] = f2bf(acc11[r]);
        } else {
            float* o = (float*)Out;
            o[(long)(m0 + m) * ldd + n0 + n]           = acc00[r];
            o[(long)(m0 + m) * ldd + n0 + 16 + n]      = acc01[r];
            o[(long)(m0 + m + 16) * ldd + n0 + n]      = acc10[r];
            o[(long)(m0 + m + 16) * ldd + n0 + 16 + n] = acc11[r];
        }
    }
}

// ---------------- LDS-staged WMMA GEMM: 128x128 block tile, double-buffered ----------------
// 256 threads = 8 waves (2 x 4). Wave tile 64x32 (4x2 WMMA frags).
template <bool BF16OUT>
__global__ __launch_bounds__(256) void gemm_lds_kernel(
    const unsigned short* __restrict__ A, int lda, long aStrideB, long aStrideH,
    const unsigned short* __restrict__ W, int ldw, long wStrideH,
    void* __restrict__ Out, int ldd, long oStrideZ,
    int K, int Hdim)
{
    __shared__ __align__(16) unsigned short As[2][128 * LDSTR];
    __shared__ __align__(16) unsigned short Bs[2][128 * LDSTR];

    int z = blockIdx.z;
    int b = z / Hdim, h = z % Hdim;
    A += (long)b * aStrideB + (long)h * aStrideH;
    W += (long)h * wStrideH;
    long mBlk = (long)blockIdx.y * 128;
    long nBlk = (long)blockIdx.x * 128;

    int tid = threadIdx.x;
    int r0 = tid >> 2;              // 0..63
    int r1 = r0 + 64;               // 64..127
    int c0 = (tid & 3) * 8;         // 0,8,16,24

    auto stage = [&](int kt, int buf) {
        int kOff = kt * 32;
        cp16(&As[buf][r0 * LDSTR + c0], A + (mBlk + r0) * lda + kOff + c0);
        cp16(&As[buf][r1 * LDSTR + c0], A + (mBlk + r1) * lda + kOff + c0);
        cp16(&Bs[buf][r0 * LDSTR + c0], W + (nBlk + r0) * ldw + kOff + c0);
        cp16(&Bs[buf][r1 * LDSTR + c0], W + (nBlk + r1) * ldw + kOff + c0);
    };

    int wave = tid >> 5;
    int wm = (wave >> 2) * 64;      // 0 or 64
    int wn = (wave & 3) * 32;       // 0,32,64,96

    f32x8 acc[4][2];
#pragma unroll
    for (int i = 0; i < 4; ++i)
#pragma unroll
        for (int j = 0; j < 2; ++j) acc[i][j] = (f32x8){};

    int nk = K / 32;
    stage(0, 0);
    for (int kt = 0; kt < nk; ++kt) {
        int cur = kt & 1;
        wait_async();
        __syncthreads();
        if (kt + 1 < nk) stage(kt + 1, cur ^ 1);

        u32x8 afr[4], bfr[2];
#pragma unroll
        for (int i = 0; i < 4; ++i) afr[i] = load_frag_a(&As[cur][(wm + i * 16) * LDSTR], LDSTR);
#pragma unroll
        for (int j = 0; j < 2; ++j) bfr[j] = load_frag_b(&Bs[cur][(wn + j * 16) * LDSTR], LDSTR);
#pragma unroll
        for (int i = 0; i < 4; ++i)
#pragma unroll
            for (int j = 0; j < 2; ++j) acc[i][j] = wmma_bf16(afr[i], bfr[j], acc[i][j]);
        __syncthreads();
    }

    int lane = tid & 31;
    int n = lane & 15, hi = lane >> 4;
    long obase = (long)z * oStrideZ;
#pragma unroll
    for (int i = 0; i < 4; ++i) {
#pragma unroll
        for (int j = 0; j < 2; ++j) {
#pragma unroll
            for (int r = 0; r < 8; ++r) {
                long m = mBlk + wm + i * 16 + r + 8 * hi;
                long nn = nBlk + wn + j * 16 + n;
                if constexpr (BF16OUT)
                    ((unsigned short*)Out)[obase + m * ldd + nn] = f2bf(acc[i][j][r]);
                else
                    ((float*)Out)[obase + m * ldd + nn] = acc[i][j][r];
            }
        }
    }
}

// ---------------- fused attention: 32 query rows / block ----------------
// grid (S/32, H, B), 128 threads (4 waves). LDS: 32 x 2048 bf16 scores (128KB).
__global__ __launch_bounds__(128) void attn_kernel(
    const unsigned short* __restrict__ qext,  // (B,H,S,576)
    const unsigned short* __restrict__ kb,    // (B,S,576)
    const unsigned short* __restrict__ ct,    // (B,C,S)
    unsigned short* __restrict__ oc,          // (B,S,H,C)
    float scale)
{
    __shared__ unsigned short sc[32 * S_];
    __shared__ float red[128];
    __shared__ float rowm[32], rowl[32];

    int b = blockIdx.z, h = blockIdx.y;
    int s0 = blockIdx.x * 32;
    int wave = threadIdx.x >> 5;
    int lane = threadIdx.x & 31;
    const unsigned short* qA = qext + ((long)(b * H_ + h) * S_ + s0) * KEXT_;
    const unsigned short* kB = kb + (long)b * S_ * KEXT_;

    // phase 1: raw scores (32 x S) -> LDS bf16, scaled
    for (int tt = wave; tt < S_ / 16; tt += 4) {
        f32x8 acc0 = {}, acc1 = {};
        for (int kk = 0; kk < KEXT_; kk += 32) {
            u32x8 a0 = load_frag_a(qA + kk, KEXT_);
            u32x8 a1 = load_frag_a(qA + (long)16 * KEXT_ + kk, KEXT_);
            u32x8 bf = load_frag_b(kB + (long)tt * 16 * KEXT_ + kk, KEXT_);
            acc0 = wmma_bf16(a0, bf, acc0);
            acc1 = wmma_bf16(a1, bf, acc1);
        }
        int n = lane & 15, hi = lane >> 4;
#pragma unroll
        for (int r = 0; r < 8; ++r) {
            sc[(r + 8 * hi) * S_ + tt * 16 + n]        = f2bf(acc0[r] * scale);
            sc[(16 + r + 8 * hi) * S_ + tt * 16 + n]   = f2bf(acc1[r] * scale);
        }
    }
    __syncthreads();

    // softmax: 4 threads per row, 512 elements each
    int row = threadIdx.x >> 2;
    int sub = threadIdx.x & 3;
    float m = -3.0e38f;
    for (int t = sub * 512; t < sub * 512 + 512; ++t)
        m = fmaxf(m, bf2f(sc[row * S_ + t]));
    red[threadIdx.x] = m; __syncthreads();
    if (sub == 0) {
        float mm = red[threadIdx.x];
        for (int j = 1; j < 4; ++j) mm = fmaxf(mm, red[threadIdx.x + j]);
        rowm[row] = mm;
    }
    __syncthreads();
    float rm = rowm[row];
    float l = 0.f;
    for (int t = sub * 512; t < sub * 512 + 512; ++t) {
        float e = __expf(bf2f(sc[row * S_ + t]) - rm);
        sc[row * S_ + t] = f2bf(e);
        l += e;
    }
    red[threadIdx.x] = l; __syncthreads();
    if (sub == 0) {
        float s = red[threadIdx.x];
        for (int j = 1; j < 4; ++j) s += red[threadIdx.x + j];
        rowl[row] = s;
    }
    __syncthreads();

    // phase 2: out_c = P (32 x S) * ct^T; wave owns 128 output cols
    f32x8 acc[2][8];
#pragma unroll
    for (int i = 0; i < 2; ++i)
#pragma unroll
        for (int j = 0; j < 8; ++j) acc[i][j] = (f32x8){};
    const unsigned short* ctb = ct + (long)b * C_ * S_;
    int mA = lane & 15, kbA = (lane >> 4) << 3;
    for (int k = 0; k < S_; k += 32) {
        u32x8 a0, a1;
        {
            const unsigned short* p0 = &sc[mA * S_ + k + kbA];
            const unsigned short* p1 = &sc[(16 + mA) * S_ + k + kbA];
#pragma unroll
            for (int j = 0; j < 4; ++j) {
                a0[j]     = *(const unsigned int*)(p0 + 2 * j);
                a0[4 + j] = *(const unsigned int*)(p0 + 16 + 2 * j);
                a1[j]     = *(const unsigned int*)(p1 + 2 * j);
                a1[4 + j] = *(const unsigned int*)(p1 + 16 + 2 * j);
            }
        }
#pragma unroll
        for (int j = 0; j < 8; ++j) {
            u32x8 bf = load_frag_b(ctb + (long)(wave * 128 + j * 16) * S_ + k, S_);
            acc[0][j] = wmma_bf16(a0, bf, acc[0][j]);
            acc[1][j] = wmma_bf16(a1, bf, acc[1][j]);
        }
    }
    int n = lane & 15, hi = lane >> 4;
#pragma unroll
    for (int j = 0; j < 8; ++j) {
#pragma unroll
        for (int r = 0; r < 8; ++r) {
            int mr = r + 8 * hi;
            long s_a = s0 + mr;
            long s_b = s0 + 16 + mr;
            float v0 = acc[0][j][r] / rowl[mr];
            float v1 = acc[1][j][r] / rowl[16 + mr];
            oc[((s_a * H_ + (long)h) + (long)b * S_ * H_) * C_ + wave * 128 + j * 16 + n] = f2bf(v0);
            oc[((s_b * H_ + (long)h) + (long)b * S_ * H_) * C_ + wave * 128 + j * 16 + n] = f2bf(v1);
        }
    }
}

// ---------------- host launch ----------------

extern "C" void kernel_launch(void* const* d_in, const int* in_sizes, int n_in,
                              void* d_out, int out_size, void* d_ws, size_t ws_size,
                              hipStream_t stream) {
    (void)in_sizes; (void)n_in; (void)out_size;
    const float* x    = (const float*)d_in[0];
    const float* fcos = (const float*)d_in[1];
    const float* fsin = (const float*)d_in[2];
    const float* wq   = (const float*)d_in[3];
    const float* wkva = (const float*)d_in[4];
    const float* wkvb = (const float*)d_in[5];
    const float* wo   = (const float*)d_in[6];
    float* out = (float*)d_out;

    auto au = [](size_t v) { return (v + 255) & ~(size_t)255; };
    char* p = (char*)d_ws;
    size_t off = 0;
    auto alloc = [&](size_t bytes) { char* r = p + off; off = au(off + bytes); return r; };

    const size_t nX   = (size_t)B_ * S_ * D_;
    const size_t nWQ  = (size_t)H_ * QK_ * D_;
    const size_t nWA  = (size_t)KEXT_ * D_;
    const size_t nWO  = (size_t)D_ * H_ * V_;
    const size_t nWB  = (size_t)H_ * 256 * C_;

    unsigned short* xb   = (unsigned short*)alloc(nX * 2);
    unsigned short* wqb  = (unsigned short*)alloc(nWQ * 2);
    unsigned short* wab  = (unsigned short*)alloc(nWA * 2);
    unsigned short* wob  = (unsigned short*)alloc(nWO * 2);
    unsigned short* wnt  = (unsigned short*)alloc((size_t)H_ * C_ * NOPE_ * 2);
    unsigned short* wvb  = (unsigned short*)alloc((size_t)H_ * V_ * C_ * 2);
    unsigned short* q    = (unsigned short*)alloc((size_t)B_ * S_ * H_ * QK_ * 2);
    float*          kvf  = (float*)alloc((size_t)B_ * S_ * KEXT_ * 4);
    unsigned short* kb   = (unsigned short*)alloc((size_t)B_ * S_ * KEXT_ * 2);
    unsigned short* ct   = (unsigned short*)alloc((size_t)B_ * C_ * S_ * 2);
    unsigned short* qext = (unsigned short*)alloc((size_t)B_ * H_ * S_ * KEXT_ * 2);
    unsigned short* oc   = (unsigned short*)alloc((size_t)B_ * S_ * H_ * C_ * 2);
    unsigned short* ob   = (unsigned short*)alloc((size_t)B_ * S_ * H_ * V_ * 2);
    if (off > ws_size) return;

    auto cvt = [&](const float* s, unsigned short* d, size_t n) {
        int blocks = (int)((n + 255) / 256); if (blocks > 8192) blocks = 8192;
        f2bf_kernel<<<blocks, 256, 0, stream>>>(s, d, (int)n);
    };
    cvt(x, xb, nX); cvt(wq, wqb, nWQ); cvt(wkva, wab, nWA); cvt(wo, wob, nWO);
    split_wkvb_kernel<<<(int)(nWB / 256), 256, 0, stream>>>(wkvb, wnt, wvb);

    const int M = B_ * S_;  // 4096

    // q = x @ wq^T (bf16 out, 4096 x 3072)
    gemm_lds_kernel<true><<<dim3((H_ * QK_) / 128, M / 128, 1), 256, 0, stream>>>(
        xb, D_, 0, 0, wqb, D_, 0, q, H_ * QK_, 0, D_, 1);

    // kv = x @ wkv_a^T (fp32 out, 4096 x 576) -- N not 128-divisible, direct kernel
    gemm_awt_kernel<false><<<dim3(KEXT_ / 64, M / 64, 1), 128, 0, stream>>>(
        xb, D_, wab, D_, kvf, KEXT_, D_);

    kv_post_kernel<<<M, 128, 0, stream>>>(kvf, fcos, fsin, kb, ct);

    {
        int n = B_ * S_ * H_ * (ROPE_ / 2);
        rope_q_kernel<<<(n + 255) / 256, 256, 0, stream>>>(q, fcos, fsin, qext);
    }

    // q_abs = q_nope @ w_nope^T (per b,h) -> qext[...,0:512]
    gemm_lds_kernel<true><<<dim3(C_ / 128, S_ / 128, B_ * H_), 256, 0, stream>>>(
        q, H_ * QK_, (long)S_ * H_ * QK_, QK_,
        wnt, NOPE_, (long)C_ * NOPE_,
        qext, KEXT_, (long)S_ * KEXT_, NOPE_, H_);

    {
        float scale = 1.0f / sqrtf((float)QK_);
        attn_kernel<<<dim3(S_ / 32, H_, B_), 128, 0, stream>>>(qext, kb, ct, oc, scale);
    }

    // out_hv = out_c @ w_v^T (per head) -> ob (B,S,H*V)
    gemm_lds_kernel<true><<<dim3(V_ / 128, M / 128, H_), 256, 0, stream>>>(
        oc, H_ * C_, 0, C_,
        wvb, C_, (long)V_ * C_,
        ob, H_ * V_, V_, C_, H_);

    // final = ob @ wo^T -> d_out (fp32)
    gemm_lds_kernel<false><<<dim3(D_ / 128, M / 128, 1), 256, 0, stream>>>(
        ob, H_ * V_, 0, 0, wob, H_ * V_, 0, out, D_, 0, H_ * V_, 1);
}
